// SceneGraphEmbedding_50337016709803
// MI455X (gfx1250) — matrix operand order repaired and checked by reading, not compiled, not run on hardware
//
#include <hip/hip_runtime.h>
#include <hip/hip_bf16.h>

#define NN 10000
#define NG 64
#define DD 256
#define MT_NODES (NN / 16)  // 625 row tiles, exact

typedef __attribute__((ext_vector_type(16))) __bf16 v16bf;
typedef __attribute__((ext_vector_type(8)))  float  v8f;
typedef __attribute__((ext_vector_type(4)))  float  v4f;

__device__ __forceinline__ void atomAddF(float* p, float v) {
  __hip_atomic_fetch_add(p, v, __ATOMIC_RELAXED, __HIP_MEMORY_SCOPE_AGENT);
}

// ---------------------------------------------------------------------------
// Convert a 256x256 f32 weight matrix (row-major [K][N]) into CDNA5 WMMA
// B-fragment layout, split into bf16 hi/lo pairs (bf16x3 trick).
// Record (nt, kb, lane): 16 bf16 hi values then 16 bf16 lo values (64 bytes).
// Lane l: N = nt*16 + (l&15), K = kb*32 + (l>>4)*16 + i, i in [0,16).
// ---------------------------------------------------------------------------
__global__ __launch_bounds__(256) void prep_w_kernel(const float* __restrict__ W,
                                                     __bf16* __restrict__ frag) {
  int idx = blockIdx.x * 256 + threadIdx.x;  // 0..4095 = (nt*8 + kb)*32 + lane
  if (idx >= 16 * 8 * 32) return;
  int lane = idx & 31;
  int kb   = (idx >> 5) & 7;
  int nt   = idx >> 8;
  int n  = nt * 16 + (lane & 15);
  int k0 = kb * 32 + (lane >> 4) * 16;
  __bf16* rec = frag + (size_t)idx * 32;
#pragma unroll
  for (int i = 0; i < 16; ++i) {
    float v = W[(size_t)(k0 + i) * DD + n];
    __bf16 hi = (__bf16)v;
    rec[i]      = hi;
    rec[16 + i] = (__bf16)(v - (float)hi);
  }
}

// ---------------------------------------------------------------------------
// C[M,256] = op(A[M,256]) @ W + bias via bf16x3 v_wmma_f32_16x16x32_bf16.
// One wave owns a 16x64 C slab (4 N-tiles): the A fragment f32->bf16 hi/lo
// conversion (the dominant VALU cost) is amortized over 4 tiles, and A is
// read 4x less often. FUSE (compile-time): bit0 = relu on A load,
// bit1 = relu on store, bit2 = add bias.
// ---------------------------------------------------------------------------
template <int FUSE>
__global__ __launch_bounds__(256) void gemm_bf16x3_kernel(
    const float* __restrict__ A, const __bf16* __restrict__ Wf,
    const float* __restrict__ bias, float* __restrict__ C, int mtiles) {
  int wave = blockIdx.x * 8 + (threadIdx.x >> 5);
  if (wave >= mtiles * 4) return;             // wave-uniform, EXEC stays full
  int lane = threadIdx.x & 31;
  int mt = wave >> 2;                         // row tile
  int ng = (wave & 3) * 4;                    // first of 4 owned N-tiles

  // A fragment addressing: lane l holds M = mt*16 + (l&15),
  // K = (l>>4)*16 .. +15 within each 32-K block.
  int row = mt * 16 + (lane & 15);
  const float*  ap  = A + (size_t)row * DD + (lane >> 4) * 16;
  const __bf16* bpl = Wf + (size_t)lane * 32;  // + ((nt*8 + kb)*32)*32

  v8f acc[4] = {};
#pragma unroll
  for (int kb = 0; kb < 8; ++kb) {
    alignas(16) float xs[16];
    *(v4f*)(xs + 0)  = *(const v4f*)(ap + 0);
    *(v4f*)(xs + 4)  = *(const v4f*)(ap + 4);
    *(v4f*)(xs + 8)  = *(const v4f*)(ap + 8);
    *(v4f*)(xs + 12) = *(const v4f*)(ap + 12);
    v16bf ahi, alo;
#pragma unroll
    for (int i = 0; i < 16; ++i) {
      float v = xs[i];
      if (FUSE & 1) v = fmaxf(v, 0.0f);
      __bf16 h = (__bf16)v;
      ahi[i] = h;
      alo[i] = (__bf16)(v - (float)h);
    }
#pragma unroll
    for (int t = 0; t < 4; ++t) {
      const __bf16* bp = bpl + ((size_t)((ng + t) * 8 + kb) * 32) * 32;
      v16bf bhi = *(const v16bf*)(bp);
      v16bf blo = *(const v16bf*)(bp + 16);
      acc[t] = __builtin_amdgcn_wmma_f32_16x16x32_bf16(
          false, ahi, false, bhi, (short)0, acc[t], false, false);
      acc[t] = __builtin_amdgcn_wmma_f32_16x16x32_bf16(
          false, ahi, false, blo, (short)0, acc[t], false, false);
      acc[t] = __builtin_amdgcn_wmma_f32_16x16x32_bf16(
          false, alo, false, bhi, (short)0, acc[t], false, false);
    }
    ap += 32;
  }

  // C/D layout: VGPR v holds M = mt*16 + 8*(l>>4) + v, N = nt*16 + (l&15)
  int m0 = mt * 16 + (lane >> 4) * 8;
#pragma unroll
  for (int t = 0; t < 4; ++t) {
    int col = (ng + t) * 16 + (lane & 15);
    float b = (FUSE & 4) ? bias[col] : 0.0f;
#pragma unroll
    for (int v = 0; v < 8; ++v) {
      float r = acc[t][v] + b;
      if (FUSE & 2) r = fmaxf(r, 0.0f);
      C[(size_t)(m0 + v) * DD + col] = r;
    }
  }
}

// --------------------------- small helper kernels ---------------------------
__global__ void fill_kernel(float* p, float v, int n) {
  int i = blockIdx.x * blockDim.x + threadIdx.x;
  if (i < n) p[i] = v;
}

__global__ void deg_count_kernel(const int* __restrict__ dst,
                                 float* __restrict__ deg, int ne) {
  int e = blockIdx.x * blockDim.x + threadIdx.x;
  if (e < ne) atomAddF(&deg[dst[e]], 1.0f);
}

__global__ void rsqrt_kernel(float* d, int n) {
  int i = blockIdx.x * blockDim.x + threadIdx.x;
  if (i < n) d[i] = rsqrtf(d[i]);
}

// out[i] = dinv[i]^2 * hw[i] + bias   (self-loop contribution + bias)
__global__ __launch_bounds__(256) void scatter_init_kernel(
    const float* __restrict__ hw, const float* __restrict__ dinv,
    const float* __restrict__ bias, float* __restrict__ out) {
  int i = blockIdx.x, c = threadIdx.x;
  float s = dinv[i];
  out[(size_t)i * DD + c] = s * s * hw[(size_t)i * DD + c] + bias[c];
}

// out[dst] += dinv[src]*dinv[dst]*hw[src] ; one wave per edge, 8 f32 per lane
__global__ __launch_bounds__(256) void scatter_edge_kernel(
    const float* __restrict__ hw, const float* __restrict__ dinv,
    const int* __restrict__ src, const int* __restrict__ dst,
    float* __restrict__ out, int ne) {
  int e = blockIdx.x * 8 + (threadIdx.x >> 5);
  if (e >= ne) return;
  int lane = threadIdx.x & 31;
  int s = src[e], d = dst[e];
  float norm = dinv[s] * dinv[d];
  const float* hp = hw + (size_t)s * DD + lane * 8;
  float*       op = out + (size_t)d * DD + lane * 8;
  v4f x0 = *(const v4f*)(hp);
  v4f x1 = *(const v4f*)(hp + 4);
#pragma unroll
  for (int j = 0; j < 4; ++j) {
    atomAddF(op + j,     norm * x0[j]);
    atomAddF(op + 4 + j, norm * x1[j]);
  }
}

__global__ __launch_bounds__(256) void pool_accum_kernel(
    const float* __restrict__ h, const int* __restrict__ batch,
    float* __restrict__ sums, float* __restrict__ cnts) {
  int i = blockIdx.x, c = threadIdx.x;
  int g = batch[i];
  atomAddF(&sums[(size_t)g * DD + c], fmaxf(h[(size_t)i * DD + c], 0.0f));
  if (c == 0) atomAddF(&cnts[g], 1.0f);
}

__global__ __launch_bounds__(256) void pool_div_kernel(
    const float* __restrict__ sums, const float* __restrict__ cnts,
    float* __restrict__ pooled) {
  int g = blockIdx.x, c = threadIdx.x;
  pooled[(size_t)g * DD + c] = sums[(size_t)g * DD + c] / fmaxf(cnts[g], 1.0f);
}

// ---------------------------------------------------------------------------
extern "C" void kernel_launch(void* const* d_in, const int* in_sizes, int n_in,
                              void* d_out, int out_size, void* d_ws,
                              size_t ws_size, hipStream_t stream) {
  (void)n_in; (void)out_size; (void)ws_size;
  const float* x     = (const float*)d_in[0];
  const int*   edge  = (const int*)d_in[1];   // [2][E]: src row then dst row
  const int*   batch = (const int*)d_in[2];
  const float* Wi = (const float*)d_in[3];
  const float* bi = (const float*)d_in[4];
  const float* W1 = (const float*)d_in[5];
  const float* b1 = (const float*)d_in[6];
  const float* W2 = (const float*)d_in[7];
  const float* b2 = (const float*)d_in[8];
  const float* Wo = (const float*)d_in[9];
  const float* bo = (const float*)d_in[10];
  const int ne = in_sizes[1] / 2;
  const int* src = edge;
  const int* dst = edge + ne;

  // workspace layout
  char* ws = (char*)d_ws;
  const size_t FRAG = (size_t)16 * 8 * 32 * 32;       // bf16 elems / matrix
  __bf16* wf   = (__bf16*)ws;                         // 4 * 256 KB = 1 MB
  float*  B0   = (float*)(ws + 4 * FRAG * sizeof(__bf16));
  float*  B1   = B0 + (size_t)NN * DD;                // 10.24 MB each
  float*  B2   = B1 + (size_t)NN * DD;
  float*  dinv = B2 + (size_t)NN * DD;                // 10000 f32 (padded)
  float*  sums = dinv + 10240;
  float*  cnts = sums + (size_t)NG * DD;
  float*  pooled = cnts + 256;

  // 4 waves (= 4x 16x64 slabs) per row tile, 8 waves per block
  const int gemm_blocks = (MT_NODES * 4 + 7) / 8;     // 313

  // 1) weight fragments (hi/lo bf16, WMMA B layout)
  prep_w_kernel<<<16, 256, 0, stream>>>(Wi, wf + 0 * FRAG);
  prep_w_kernel<<<16, 256, 0, stream>>>(W1, wf + 1 * FRAG);
  prep_w_kernel<<<16, 256, 0, stream>>>(W2, wf + 2 * FRAG);
  prep_w_kernel<<<16, 256, 0, stream>>>(Wo, wf + 3 * FRAG);

  // 2) deg (self-loop = 1) -> dinv = rsqrt(deg)
  fill_kernel<<<(NN + 255) / 256, 256, 0, stream>>>(dinv, 1.0f, NN);
  deg_count_kernel<<<(ne + 255) / 256, 256, 0, stream>>>(dst, dinv, ne);
  rsqrt_kernel<<<(NN + 255) / 256, 256, 0, stream>>>(dinv, NN);

  // 3) h0 = relu(x @ Wi + bi)
  gemm_bf16x3_kernel<2 | 4><<<gemm_blocks, 256, 0, stream>>>(
      x, wf + 0 * FRAG, bi, B0, MT_NODES);
  // 4) conv1: hw = h0 @ W1 ; out1 = selfloop + bias ; edge scatter
  gemm_bf16x3_kernel<0><<<gemm_blocks, 256, 0, stream>>>(
      B0, wf + 1 * FRAG, nullptr, B1, MT_NODES);
  scatter_init_kernel<<<NN, 256, 0, stream>>>(B1, dinv, b1, B2);
  scatter_edge_kernel<<<(ne + 7) / 8, 256, 0, stream>>>(B1, dinv, src, dst, B2, ne);

  // 5) conv2: hw = relu(out1) @ W2 (relu fused into A conversion)
  gemm_bf16x3_kernel<1><<<gemm_blocks, 256, 0, stream>>>(
      B2, wf + 2 * FRAG, nullptr, B1, MT_NODES);
  scatter_init_kernel<<<NN, 256, 0, stream>>>(B1, dinv, b2, B0);
  scatter_edge_kernel<<<(ne + 7) / 8, 256, 0, stream>>>(B1, dinv, src, dst, B0, ne);

  // 6) global mean pool (relu fused into read)
  fill_kernel<<<(NG * DD + 255) / 256, 256, 0, stream>>>(sums, 0.0f, NG * DD);
  fill_kernel<<<1, 256, 0, stream>>>(cnts, 0.0f, NG);
  pool_accum_kernel<<<NN, 256, 0, stream>>>(B0, batch, sums, cnts);
  pool_div_kernel<<<NG, 256, 0, stream>>>(sums, cnts, pooled);

  // 7) out = pooled @ Wo + bo   (64 rows = 4 row tiles -> 16 waves)
  gemm_bf16x3_kernel<4><<<2, 256, 0, stream>>>(
      pooled, wf + 3 * FRAG, bo, (float*)d_out, 4);
}
